// Block_88261577932905
// MI455X (gfx1250) — compile-verified, matrix operand order checked by reading, not run on hardware
//
#include <hip/hip_runtime.h>
#include <math.h>

// ---------------------------------------------------------------------------
// Transformer block for MI455X (gfx1250, wave32, WMMA bf16 16x16x32, f32 acc)
// Data movement: TDM tensor_load_to_lds for row-major tiles (+ s_wait_tensorcnt),
// manual transpose-scatter for tiles that must land transposed in LDS.
// ---------------------------------------------------------------------------

typedef __bf16 bf16;
typedef __attribute__((ext_vector_type(4)))  bf16  bf16x4;
typedef __attribute__((ext_vector_type(8)))  bf16  bf16x8;
typedef __attribute__((ext_vector_type(16))) bf16  v16bf;
typedef __attribute__((ext_vector_type(8)))  float v8f;
typedef __attribute__((ext_vector_type(4)))  unsigned int v4u;
typedef __attribute__((ext_vector_type(8)))  int   v8i;
typedef __attribute__((ext_vector_type(4)))  int   v4i;

#define WMMA_BF16(a, b, c) \
  __builtin_amdgcn_wmma_f32_16x16x32_bf16(false, (a), false, (b), (short)0, (c), false, false)

#if __has_builtin(__builtin_amdgcn_tensor_load_to_lds)
#define HAVE_TDM 1
#else
#define HAVE_TDM 0
#endif

#if HAVE_TDM
// 2-D tile DMA: global (row-major, line stride in elements) -> LDS (packed rows).
// Descriptor packing per CDNA5 ISA ch.8 (D# groups 0/1); groups 2/3 zero (<=2D).
static __device__ __forceinline__ void tdm_load_2d(const bf16* gptr, unsigned lds_addr,
                                                   unsigned tile_w, unsigned tile_h,
                                                   unsigned stride_elems) {
  unsigned long long ga = (unsigned long long)(uintptr_t)gptr;
  v4u g0;
  g0[0] = 1u;                                   // count=1, user load descriptor
  g0[1] = lds_addr;                             // LDS byte address
  g0[2] = (unsigned)ga;                         // global_addr[31:0]
  g0[3] = (unsigned)((ga >> 32) & 0x01FFFFFFu)  // global_addr[56:32]
        | (2u << 30);                           // type=2 ("image")
  v8i g1;
  g1[0] = (int)(1u << 16);                      // workgroup_mask=0, data_size=1 (2B)
  g1[1] = (int)(tile_w << 16);                  // tensor_dim0[15:0]  (== tile_w)
  g1[2] = (int)(tile_h << 16);                  // tensor_dim0 hi=0, tensor_dim1[15:0]
  g1[3] = (int)(tile_w << 16);                  // tensor_dim1 hi=0, tile_dim0
  g1[4] = (int)(tile_h & 0xFFFFu);              // tile_dim1; tile_dim2=0 (2D)
  g1[5] = (int)stride_elems;                    // tensor_dim0_stride[31:0]
  g1[6] = 0;                                    // stride0 hi, stride1 lo
  g1[7] = 0;
  v4i z = {0, 0, 0, 0};
#if __clang_major__ >= 23
  v8i z8 = {0, 0, 0, 0, 0, 0, 0, 0};
  __builtin_amdgcn_tensor_load_to_lds(g0, g1, z, z, z8, 0);
#else
  __builtin_amdgcn_tensor_load_to_lds(g0, g1, z, z, 0);
#endif
}
#endif

static __device__ __forceinline__ unsigned lds_off(const void* p) {
  // Flat shared-aperture address: low 32 bits are the LDS byte offset (ISA 10.2).
  return (unsigned)(uintptr_t)p;
}

static __device__ __forceinline__ v16bf pack16(bf16x8 lo, bf16x8 hi) {
  v16bf r;
#pragma unroll
  for (int j = 0; j < 8; ++j) { r[j] = lo[j]; r[j + 8] = hi[j]; }
  return r;
}
// 16-bit A/B fragment: elements 0..7 at p, 8..15 at p+16 elements
// (CDNA5 16-bit matrix VGPR layout; p includes row*ld + khalf, khalf=(lane>=16)*8).
static __device__ __forceinline__ v16bf frag_ld(const bf16* p) {
  bf16x8 lo = *(const bf16x8*)p;
  bf16x8 hi = *(const bf16x8*)(p + 16);
  return pack16(lo, hi);
}

// ---------------------------------------------------------------------------
// Weight preparation (fp32 -> bf16, QKV column-concat repack)
// ---------------------------------------------------------------------------
__global__ __launch_bounds__(256) void cast_f32_to_bf16(const float* __restrict__ in,
                                                        bf16* __restrict__ out, int n) {
  int i = blockIdx.x * 256 + threadIdx.x;
  if (i < n) out[i] = (bf16)in[i];
}

// out[e*3072 + which*1024 + h*64 + d] = W{q,k,v}[h][e][d]   (W is [H=16, E=1024, DH=64])
__global__ __launch_bounds__(256) void pack_qkv_w(const float* __restrict__ Wq,
                                                  const float* __restrict__ Wk,
                                                  const float* __restrict__ Wv,
                                                  bf16* __restrict__ out) {
  int i = blockIdx.x * 256 + threadIdx.x;   // over 3*1024*1024
  int which = i >> 20;
  int r = i & 0xFFFFF;
  int e = r >> 10;
  int c = r & 1023;                          // h*64 + d
  const float* W = (which == 0) ? Wq : (which == 1) ? Wk : Wv;
  float v = W[(size_t)(c >> 6) * (1024 * 64) + (size_t)e * 64 + (c & 63)];
  out[(size_t)e * 3072 + (size_t)which * 1024 + c] = (bf16)v;
}

// ---------------------------------------------------------------------------
// LayerNorm (fp32 in) -> bf16 out.  One block per row, E = 1024.
// ---------------------------------------------------------------------------
__global__ __launch_bounds__(256) void ln_to_bf16(const float* __restrict__ X,
                                                  const float* __restrict__ g,
                                                  const float* __restrict__ bb,
                                                  bf16* __restrict__ out) {
  constexpr int E = 1024;
  int row = blockIdx.x, t = threadIdx.x;
  const float* x = X + (size_t)row * E;
  float4 v = *(const float4*)(x + t * 4);
  float s  = v.x + v.y + v.z + v.w;
  float ss = v.x * v.x + v.y * v.y + v.z * v.z + v.w * v.w;
#pragma unroll
  for (int m = 16; m; m >>= 1) { s += __shfl_xor(s, m, 32); ss += __shfl_xor(ss, m, 32); }
  __shared__ float rs[8], rq[8];
  int wid = t >> 5, lane = t & 31;
  if (lane == 0) { rs[wid] = s; rq[wid] = ss; }
  __syncthreads();
  if (t == 0) {
    float a = 0.f, b2 = 0.f;
#pragma unroll
    for (int i = 0; i < 8; ++i) { a += rs[i]; b2 += rq[i]; }
    rs[0] = a; rq[0] = b2;
  }
  __syncthreads();
  float mean = rs[0] * (1.0f / E);
  float var  = rq[0] * (1.0f / E) - mean * mean;
  float inv  = rsqrtf(var + 1e-5f);
  float4 gg = *(const float4*)(g  + t * 4);
  float4 bv = *(const float4*)(bb + t * 4);
  bf16x4 o;
  o[0] = (bf16)((v.x - mean) * inv * gg.x + bv.x);
  o[1] = (bf16)((v.y - mean) * inv * gg.y + bv.y);
  o[2] = (bf16)((v.z - mean) * inv * gg.z + bv.z);
  o[3] = (bf16)((v.w - mean) * inv * gg.w + bv.w);
  *(bf16x4*)(out + (size_t)row * E + t * 4) = o;
}

// ---------------------------------------------------------------------------
// Generic bf16 GEMM: C[M,N] = A[M,K] * B[K,N], f32 accumulate.
// EPI 0: store bf16.  EPI 1: += residual R (f32), store f32.  EPI 2: GELU, store bf16.
// Block 256 thr (8 waves), tile 128x128, BK=32; wave computes 32x64 (8 WMMA / K-step).
// A tile staged via TDM (tensor_load_to_lds); B tile transpose-scattered manually.
// ---------------------------------------------------------------------------
template <int EPI>
__global__ __launch_bounds__(256) void gemm_bf16(const bf16* __restrict__ A,
                                                 const bf16* __restrict__ B,
                                                 const float* __restrict__ R,
                                                 void* __restrict__ Cout,
                                                 int M, int N, int K) {
  constexpr int BM = 128, BN = 128, BK = 32;
  __shared__ bf16 sA[BM * BK];    // row-major [m][k]
  __shared__ bf16 sBt[BN * BK];   // transposed [n][k]
  const int bm = blockIdx.y * BM, bn = blockIdx.x * BN;
  const int t = threadIdx.x, wid = t >> 5, lane = t & 31;
  const int wm = (wid & 3) * 32;      // 4 waves along M
  const int wn = (wid >> 2) * 64;     // 2 waves along N
  const int lm = lane & 15;
  const int kh = (lane >> 4) * 8;     // K half offset per ISA layout
#if HAVE_TDM
  const unsigned sA_addr = lds_off(sA);
#endif

  v8f acc[2][4] = {};

  for (int k0 = 0; k0 < K; k0 += BK) {
    __syncthreads();
#if HAVE_TDM
    if (wid == 0)                     // one TDM op per tile per block
      tdm_load_2d(A + (size_t)bm * K + k0, sA_addr, BK, BM, K);
#else
#pragma unroll
    for (int i = 0; i < 2; ++i) {     // A tile: 128x32 bf16 = 8KB
      int o = (t + i * 256) * 8;
      int r = o >> 5, c = o & 31;
      *(bf16x8*)(sA + o) = *(const bf16x8*)(A + (size_t)(bm + r) * K + k0 + c);
    }
#endif
#pragma unroll
    for (int i = 0; i < 2; ++i) {     // B tile: 32x128, store transposed
      int o = (t + i * 256) * 8;
      int kr = o >> 7, nc = o & 127;
      bf16x8 v = *(const bf16x8*)(B + (size_t)(k0 + kr) * N + bn + nc);
#pragma unroll
      for (int j = 0; j < 8; ++j) sBt[(nc + j) * BK + kr] = v[j];
      if (k0 + BK < K)
        __builtin_prefetch(B + (size_t)(k0 + BK + kr) * N + bn + nc, 0, 1);
    }
#if HAVE_TDM
    if (wid == 0) __builtin_amdgcn_s_wait_tensorcnt(0);
#endif
    __syncthreads();

    v16bf af[2], bfr[4];
#pragma unroll
    for (int mt = 0; mt < 2; ++mt)
      af[mt] = frag_ld(sA + (wm + mt * 16 + lm) * BK + kh);
#pragma unroll
    for (int nt = 0; nt < 4; ++nt)
      bfr[nt] = frag_ld(sBt + (wn + nt * 16 + lm) * BK + kh);
#pragma unroll
    for (int mt = 0; mt < 2; ++mt)
#pragma unroll
      for (int nt = 0; nt < 4; ++nt)
        acc[mt][nt] = WMMA_BF16(af[mt], bfr[nt], acc[mt][nt]);
  }

  // Epilogue: element e of a 16x16 f32 C tile -> row = e + 8*(lane>=16), col = lane&15
#pragma unroll
  for (int mt = 0; mt < 2; ++mt)
#pragma unroll
    for (int nt = 0; nt < 4; ++nt)
#pragma unroll
      for (int e = 0; e < 8; ++e) {
        int row = bm + wm + mt * 16 + e + ((lane >> 4) << 3);
        int col = bn + wn + nt * 16 + lm;
        size_t idx = (size_t)row * N + col;
        float v = acc[mt][nt][e];
        if (EPI == 1) {
          ((float*)Cout)[idx] = v + R[idx];
        } else if (EPI == 2) {
          float ge = 0.5f * v * (1.0f + erff(v * 0.70710678118654752f));
          ((bf16*)Cout)[idx] = (bf16)ge;
        } else {
          ((bf16*)Cout)[idx] = (bf16)v;
        }
      }
}

// ---------------------------------------------------------------------------
// Flash-style causal attention over packed QKV [B*S, 3072] (bf16).
// Grid: (S/256, H, B).  Block = 8 waves; wave owns 32 query rows.
// K tile staged via TDM; V transpose-scattered; P bounced via per-wave LDS.
// ---------------------------------------------------------------------------
__global__ __launch_bounds__(256) void attn_kernel(const bf16* __restrict__ QKV,
                                                   bf16* __restrict__ ctx) {
  constexpr int S = 2048, LD = 3072, DH = 64;
  const int b = blockIdx.z, hh = blockIdx.y;
  const int qbase = blockIdx.x * 256;
  const size_t rowbase = (size_t)b * S;
  const bf16* Qp = QKV + hh * DH;
  const bf16* Kp = QKV + 1024 + hh * DH;
  const bf16* Vp = QKV + 2048 + hh * DH;

  __shared__ bf16 sK[64 * 64];       // [key][d]  == B^T layout for Q*K^T
  __shared__ bf16 sVt[64 * 64];      // [d][key]  == B^T layout for P*V
  __shared__ bf16 sP[8 * 32 * 64];   // per-wave P tile (C-layout -> A-layout bounce)

  const int t = threadIdx.x, wid = t >> 5, lane = t & 31;
  const int lm = lane & 15, kh = (lane >> 4) * 8;
  const int wq = qbase + wid * 32;   // wave's first query row (within sequence)
  bf16* sPw = sP + wid * (32 * 64);
#if HAVE_TDM
  const unsigned sK_addr = lds_off(sK);
#endif

  // Q fragments straight from global (rows strided by LD) -- stay in registers.
  v16bf qf[2][2];
#pragma unroll
  for (int mt = 0; mt < 2; ++mt)
#pragma unroll
    for (int ks = 0; ks < 2; ++ks)
      qf[mt][ks] = frag_ld(Qp + (rowbase + wq + mt * 16 + lm) * LD + ks * 32 + kh);

  v8f o[2][4] = {};
  float mrow[2][8], lrow[2][8];
#pragma unroll
  for (int mt = 0; mt < 2; ++mt)
#pragma unroll
    for (int e = 0; e < 8; ++e) { mrow[mt][e] = -1e30f; lrow[mt][e] = 0.f; }

  const int kend = qbase + 256;   // causal: keys <= max query row in block
  for (int kt = 0; kt < kend; kt += 64) {
    __syncthreads();
#if HAVE_TDM
    if (wid == 0)                     // K tile 64x64 -> LDS row-major (already B^T)
      tdm_load_2d(Kp + (rowbase + kt) * LD, sK_addr, 64, 64, LD);
#else
#pragma unroll
    for (int i = 0; i < 2; ++i) {
      int oo = (t + i * 256) * 8;
      int kr = oo >> 6, c = oo & 63;
      *(bf16x8*)(sK + oo) = *(const bf16x8*)(Kp + (rowbase + kt + kr) * LD + c);
    }
#endif
#pragma unroll
    for (int i = 0; i < 2; ++i) {     // V tile 64x64 -> LDS transposed [d][key]
      int oo = (t + i * 256) * 8;
      int kr = oo >> 6, c = oo & 63;
      bf16x8 v = *(const bf16x8*)(Vp + (rowbase + kt + kr) * LD + c);
#pragma unroll
      for (int j = 0; j < 8; ++j) sVt[(c + j) * 64 + kr] = v[j];
    }
#if HAVE_TDM
    if (wid == 0) __builtin_amdgcn_s_wait_tensorcnt(0);
#endif
    __syncthreads();

    // S = Q * K^T  (32x64 scores per wave)
    v8f s[2][4] = {};
#pragma unroll
    for (int ks = 0; ks < 2; ++ks) {
      v16bf kf[4];
#pragma unroll
      for (int nt = 0; nt < 4; ++nt)
        kf[nt] = frag_ld(sK + (nt * 16 + lm) * 64 + ks * 32 + kh);
#pragma unroll
      for (int mt = 0; mt < 2; ++mt)
#pragma unroll
        for (int nt = 0; nt < 4; ++nt)
          s[mt][nt] = WMMA_BF16(qf[mt][ks], kf[nt], s[mt][nt]);
    }

    // Online softmax (row = e + 8*(lane>=16); 16-lane row reductions)
#pragma unroll
    for (int mt = 0; mt < 2; ++mt)
#pragma unroll
      for (int e = 0; e < 8; ++e) {
        int rloc = mt * 16 + e + ((lane >> 4) << 3);
        int row = wq + rloc;
        float vals[4];
        float mx = -1e30f;
#pragma unroll
        for (int nt = 0; nt < 4; ++nt) {
          int col = kt + nt * 16 + lm;
          float sv = s[mt][nt][e] * 0.125f;          // 1/sqrt(64)
          sv = (col <= row) ? sv : -1e30f;
          vals[nt] = sv;
          mx = fmaxf(mx, sv);
        }
#pragma unroll
        for (int m2 = 8; m2; m2 >>= 1) mx = fmaxf(mx, __shfl_xor(mx, m2, 16));
        float mnew = fmaxf(mrow[mt][e], mx);
        float alpha = __expf(mrow[mt][e] - mnew);
        float rs0 = 0.f;
#pragma unroll
        for (int nt = 0; nt < 4; ++nt) {
          float p = __expf(vals[nt] - mnew);
          rs0 += p;
          sPw[rloc * 64 + nt * 16 + lm] = (bf16)p;
        }
#pragma unroll
        for (int m2 = 8; m2; m2 >>= 1) rs0 += __shfl_xor(rs0, m2, 16);
        lrow[mt][e] = lrow[mt][e] * alpha + rs0;
        mrow[mt][e] = mnew;
#pragma unroll
        for (int nt = 0; nt < 4; ++nt) o[mt][nt][e] *= alpha;
      }

    // O += P * V   (P read back from per-wave LDS in A-fragment layout)
#pragma unroll
    for (int ks = 0; ks < 2; ++ks) {
      v16bf pf[2], vf[4];
#pragma unroll
      for (int mt = 0; mt < 2; ++mt)
        pf[mt] = frag_ld(sPw + (mt * 16 + lm) * 64 + ks * 32 + kh);
#pragma unroll
      for (int nt = 0; nt < 4; ++nt)
        vf[nt] = frag_ld(sVt + (nt * 16 + lm) * 64 + ks * 32 + kh);
#pragma unroll
      for (int mt = 0; mt < 2; ++mt)
#pragma unroll
        for (int nt = 0; nt < 4; ++nt)
          o[mt][nt] = WMMA_BF16(pf[mt], vf[nt], o[mt][nt]);
    }
  }

  // Normalize and emit ctx (bf16), concat-heads layout [row, h*64 + d]
#pragma unroll
  for (int mt = 0; mt < 2; ++mt)
#pragma unroll
    for (int nt = 0; nt < 4; ++nt)
#pragma unroll
      for (int e = 0; e < 8; ++e) {
        int row = wq + mt * 16 + e + ((lane >> 4) << 3);
        int col = nt * 16 + lm;
        float v = o[mt][nt][e] / lrow[mt][e];
        ctx[(rowbase + row) * 1024 + hh * DH + col] = (bf16)v;
      }
}

// ---------------------------------------------------------------------------
// Host-side orchestration
// ---------------------------------------------------------------------------
extern "C" void kernel_launch(void* const* d_in, const int* in_sizes, int n_in,
                              void* d_out, int out_size, void* d_ws, size_t ws_size,
                              hipStream_t stream) {
  (void)in_sizes; (void)n_in; (void)out_size; (void)ws_size;
  constexpr int E = 1024, EXPE = 6144, NQKV = 3072, BS = 4 * 2048;

  const float* x     = (const float*)d_in[0];
  const float* Wq    = (const float*)d_in[1];
  const float* Wk    = (const float*)d_in[2];
  const float* Wv    = (const float*)d_in[3];
  const float* Wo    = (const float*)d_in[4];
  const float* W1    = (const float*)d_in[5];
  const float* W2    = (const float*)d_in[6];
  const float* ln1_g = (const float*)d_in[7];
  const float* ln1_b = (const float*)d_in[8];
  const float* ln2_g = (const float*)d_in[9];
  const float* ln2_b = (const float*)d_in[10];
  float* out = (float*)d_out;

  char* ws = (char*)d_ws;
  size_t off = 0;
  auto alloc = [&](size_t bytes) { size_t o = off; off += (bytes + 255) & ~(size_t)255; return o; };
  bf16*  h_bf    = (bf16*)(ws + alloc((size_t)BS * E * 2));
  bf16*  qkv_bf  = (bf16*)(ws + alloc((size_t)BS * NQKV * 2));
  bf16*  ctx_bf  = (bf16*)(ws + alloc((size_t)BS * E * 2));
  float* x1      = (float*)(ws + alloc((size_t)BS * E * 4));
  bf16*  h2_bf   = (bf16*)(ws + alloc((size_t)BS * E * 2));
  bf16*  act_bf  = (bf16*)(ws + alloc((size_t)BS * EXPE * 2));
  bf16*  wqkv_bf = (bf16*)(ws + alloc((size_t)E * NQKV * 2));
  bf16*  wo_bf   = (bf16*)(ws + alloc((size_t)E * E * 2));
  bf16*  w1_bf   = (bf16*)(ws + alloc((size_t)E * EXPE * 2));
  bf16*  w2_bf   = (bf16*)(ws + alloc((size_t)EXPE * E * 2));

  // Weight conversion
  pack_qkv_w<<<3 * E * E / 256, 256, 0, stream>>>(Wq, Wk, Wv, wqkv_bf);
  cast_f32_to_bf16<<<E * E / 256, 256, 0, stream>>>(Wo, wo_bf, E * E);
  cast_f32_to_bf16<<<E * EXPE / 256, 256, 0, stream>>>(W1, w1_bf, E * EXPE);
  cast_f32_to_bf16<<<EXPE * E / 256, 256, 0, stream>>>(W2, w2_bf, EXPE * E);

  // h = LN1(x)
  ln_to_bf16<<<BS, 256, 0, stream>>>(x, ln1_g, ln1_b, h_bf);
  // QKV = h @ Wqkv
  gemm_bf16<0><<<dim3(NQKV / 128, BS / 128), 256, 0, stream>>>(h_bf, wqkv_bf, nullptr,
                                                               qkv_bf, BS, NQKV, E);
  // ctx = attention(Q,K,V)
  attn_kernel<<<dim3(2048 / 256, 16, 4), 256, 0, stream>>>(qkv_bf, ctx_bf);
  // x1 = x + ctx @ Wo
  gemm_bf16<1><<<dim3(E / 128, BS / 128), 256, 0, stream>>>(ctx_bf, wo_bf, x, x1, BS, E, E);
  // h2 = LN2(x1)
  ln_to_bf16<<<BS, 256, 0, stream>>>(x1, ln2_g, ln2_b, h2_bf);
  // act = gelu(h2 @ W1)
  gemm_bf16<2><<<dim3(EXPE / 128, BS / 128), 256, 0, stream>>>(h2_bf, w1_bf, nullptr,
                                                               act_bf, BS, EXPE, E);
  // out = x1 + act @ W2
  gemm_bf16<1><<<dim3(E / 128, BS / 128), 256, 0, stream>>>(act_bf, w2_bf, x1, out, BS, E, EXPE);
}
